// Attention_5669356830938
// MI455X (gfx1250) — compile-verified, matrix operand order checked by reading, not run on hardware
//
#include <hip/hip_runtime.h>
#include <cstdint>
#include <cstddef>

typedef __attribute__((ext_vector_type(16))) _Float16 v16h;
typedef __attribute__((ext_vector_type(8)))  _Float16 v8h;
typedef __attribute__((ext_vector_type(8)))  float    v8f;

#define NB    4
#define NTOK  4096
#define NC    768
#define NH    8
#define NHD   96
#define MTOK  (NB * NTOK)        // 16384 rows
#define SCALE 0.10206207261596575f  // 96^-0.5
#define EPS_N 1e-12f

__device__ __forceinline__ v8f v8f_zero() {
  v8f z = {0.f, 0.f, 0.f, 0.f, 0.f, 0.f, 0.f, 0.f};
  return z;
}

// pack two f16 into one dword for 32-bit LDS stores
__device__ __forceinline__ uint32_t pkh(_Float16 a, _Float16 b) {
  union { _Float16 h[2]; uint32_t u; } x;
  x.h[0] = a; x.h[1] = b;
  return x.u;
}

// Load one WMMA f16 fragment (A or B) from row-major storage.
// p0 must point at  base + row*ld + (lane>>4)*8  (16B aligned, ld % 8 == 0).
__device__ __forceinline__ v16h ld_frag(const _Float16* p0) {
  v8h lo = *(const v8h*)(p0);
  v8h hi = *(const v8h*)(p0 + 16);
  return __builtin_shufflevector(lo, hi, 0, 1, 2, 3, 4, 5, 6, 7,
                                         8, 9, 10, 11, 12, 13, 14, 15);
}

__device__ __forceinline__ v8f wmma16(v16h a, v16h b, v8f c) {
  return __builtin_amdgcn_wmma_f32_16x16x32_f16(false, a, false, b,
                                                (short)0, c, false, false);
}

// ---------------------------------------------------------------------------
// Generic tiled GEMM:  C[M,N] = A[M,K] (f32 or f16) * W[K,N] (f32, cvt->f16)
// Block tile 128x128, BK=32, 8 waves, each wave computes 64x32 (4x2 tiles).
// Double-buffered LDS: global loads for tile k+1 overlap WMMA on tile k.
// ---------------------------------------------------------------------------
template <bool A16, bool BIAS>
__global__ void __launch_bounds__(256)
gemm_kernel(const void* __restrict__ Aptr, const float* __restrict__ W,
            void* __restrict__ Cptr, const float* __restrict__ bias,
            int M, int N, int K) {
  __shared__ _Float16 As[2][128 * 40];   // [m][k], ld=40 (mult of 8)
  __shared__ _Float16 Bs[2][128 * 40];   // [n][k], ld=40

  const int t = threadIdx.x;
  const int lane = t & 31;
  const int wave = t >> 5;     // 0..7
  const int wm = wave >> 2;    // 0..1  (row half)
  const int wn = wave & 3;     // 0..3  (col quarter)
  const int bm = blockIdx.y * 128;
  const int bn = blockIdx.x * 128;

  v8f acc[4][2];
#pragma unroll
  for (int i = 0; i < 4; i++)
#pragma unroll
    for (int j = 0; j < 2; j++) acc[i][j] = v8f_zero();

  const int ar = t >> 1;            // A stage: row 0..127
  const int akc = (t & 1) * 16;     // A stage: k sub 0/16
  const int wkr = (t >> 4) * 2;     // W stage: k row pair 0,2,..,30
  const int wnc = (t & 15) * 8;     // W stage: n sub 0..120

  float4 xa[4];                     // staged A (f32 path)
  v8h    ha[2];                     // staged A (f16 path)
  float4 w0a, w0b, w1a, w1b;        // staged W (two k-rows x 8 n)

  // ---- stage loaders / committers ----
  auto load_tile = [&](int k0) {
    if (A16) {
      const _Float16* ga = (const _Float16*)Aptr + (size_t)(bm + ar) * K + k0 + akc;
      ha[0] = *(const v8h*)ga;
      ha[1] = *(const v8h*)(ga + 8);
    } else {
      const float* ga = (const float*)Aptr + (size_t)(bm + ar) * K + k0 + akc;
#pragma unroll
      for (int q = 0; q < 4; q++) xa[q] = ((const float4*)ga)[q];
    }
    const float* gw = W + (size_t)(k0 + wkr) * N + bn + wnc;
    w0a = ((const float4*)gw)[0];
    w0b = ((const float4*)gw)[1];
    w1a = ((const float4*)(gw + N))[0];
    w1b = ((const float4*)(gw + N))[1];
  };

  auto commit_tile = [&](int buf) {
    _Float16* as = As[buf];
    _Float16* bs = Bs[buf];
    if (A16) {
      *(v8h*)&as[ar * 40 + akc]     = ha[0];
      *(v8h*)&as[ar * 40 + akc + 8] = ha[1];
    } else {
#pragma unroll
      for (int q = 0; q < 4; q++) {
        *(uint32_t*)&as[ar * 40 + akc + q * 4]     = pkh((_Float16)xa[q].x, (_Float16)xa[q].y);
        *(uint32_t*)&as[ar * 40 + akc + q * 4 + 2] = pkh((_Float16)xa[q].z, (_Float16)xa[q].w);
      }
    }
    float r0[8] = {w0a.x, w0a.y, w0a.z, w0a.w, w0b.x, w0b.y, w0b.z, w0b.w};
    float r1[8] = {w1a.x, w1a.y, w1a.z, w1a.w, w1b.x, w1b.y, w1b.z, w1b.w};
#pragma unroll
    for (int j = 0; j < 8; j++)
      *(uint32_t*)&bs[(wnc + j) * 40 + wkr] = pkh((_Float16)r0[j], (_Float16)r1[j]);
  };

  // ---- prologue: stage tile 0 ----
  load_tile(0);
  commit_tile(0);
  __syncthreads();

  int cur = 0;
  for (int k0 = 0; k0 < K; k0 += 32) {
    const bool more = (k0 + 32) < K;
    if (more) load_tile(k0 + 32);      // global loads overlap WMMA below

    v16h a[4], bf[2];
#pragma unroll
    for (int i = 0; i < 4; i++)
      a[i] = ld_frag(&As[cur][(wm * 64 + i * 16 + (lane & 15)) * 40 + (lane >> 4) * 8]);
#pragma unroll
    for (int j = 0; j < 2; j++)
      bf[j] = ld_frag(&Bs[cur][(wn * 32 + j * 16 + (lane & 15)) * 40 + (lane >> 4) * 8]);
#pragma unroll
    for (int i = 0; i < 4; i++)
#pragma unroll
      for (int j = 0; j < 2; j++) acc[i][j] = wmma16(a[i], bf[j], acc[i][j]);

    if (more) commit_tile(cur ^ 1);    // write other buffer (no read conflict)
    __syncthreads();
    cur ^= 1;
  }

#pragma unroll
  for (int i = 0; i < 4; i++) {
#pragma unroll
    for (int j = 0; j < 2; j++) {
      int row0 = bm + wm * 64 + i * 16 + (lane >> 4) * 8;
      int col = bn + wn * 32 + j * 16 + (lane & 15);
#pragma unroll
      for (int e = 0; e < 8; e++) {
        int row = row0 + e;
        if (BIAS)
          ((float*)Cptr)[(size_t)row * N + col] = acc[i][j][e] + bias[col];
        else
          ((_Float16*)Cptr)[(size_t)row * N + col] = (_Float16)acc[i][j][e];
      }
    }
  }
}

// ---------------------------------------------------------------------------
// Per-channel sum of squares over tokens (for L2 norms over the token axis).
// grid (B, C/128, 16 chunks), block 128; atomicAdd partials.
// ---------------------------------------------------------------------------
__global__ void sumsq_kernel(const _Float16* __restrict__ X, int rowStride,
                             float* __restrict__ sums) {
  int b = blockIdx.x;
  int c = blockIdx.y * 128 + threadIdx.x;
  int n0 = blockIdx.z * 256;
  const _Float16* p = X + ((size_t)b * NTOK + n0) * rowStride + c;
  float s = 0.f;
  for (int n = 0; n < 256; n++) {
    float v = (float)p[(size_t)n * rowStride];
    s += v * v;
  }
  atomicAdd(&sums[b * NC + c], s);
}

__global__ void invnorm_kernel(float* __restrict__ sums, int n) {
  int i = blockIdx.x * blockDim.x + threadIdx.x;
  if (i < n) sums[i] = 1.f / fmaxf(sqrtf(sums[i]), EPS_N);
}

// ---------------------------------------------------------------------------
// S[bh][d][e] = sum_n Q[b][n][off+d] * K[b][n][off+e]   (K-dim = 4096 tokens)
// One block per (b,h): 6 waves, wave w owns rows d in [16w,16w+16), 6 col tiles.
// Stage 64-token chunks of Q/K transposed into LDS ([ch][tok], ld=72).
// Each thread stages a token *pair* so transpose stores are packed b32.
// ---------------------------------------------------------------------------
__global__ void __launch_bounds__(192)
s_kernel(const _Float16* __restrict__ Q, const _Float16* __restrict__ KV,
         float* __restrict__ S) {
  __shared__ _Float16 Qs[96 * 72];
  __shared__ _Float16 Ks[96 * 72];
  int bh = blockIdx.x;
  int b = bh >> 3, h = bh & 7;
  int off = h * NHD;
  int t = threadIdx.x, lane = t & 31, wave = t >> 5;  // 6 waves

  v8f acc[6];
#pragma unroll
  for (int j = 0; j < 6; j++) acc[j] = v8f_zero();

  const int np2 = (t & 31) * 2;    // token pair base 0..62
  const int c0 = (t >> 5) * 16;    // channel group 0,16,...,80

  for (int n0 = 0; n0 < NTOK; n0 += 64) {
    const _Float16* gq0 = Q + ((size_t)b * NTOK + n0 + np2) * NC + off + c0;
    const _Float16* gk0 = KV + ((size_t)b * NTOK + n0 + np2) * (2 * NC) + off + c0;
    v8h qa0 = ((const v8h*)gq0)[0], qa1 = ((const v8h*)gq0)[1];
    v8h qb0 = ((const v8h*)(gq0 + NC))[0], qb1 = ((const v8h*)(gq0 + NC))[1];
    v8h ka0 = ((const v8h*)gk0)[0], ka1 = ((const v8h*)gk0)[1];
    v8h kb0 = ((const v8h*)(gk0 + 2 * NC))[0], kb1 = ((const v8h*)(gk0 + 2 * NC))[1];
#pragma unroll
    for (int j = 0; j < 8; j++) {
      *(uint32_t*)&Qs[(c0 + j) * 72 + np2]     = pkh(qa0[j], qb0[j]);
      *(uint32_t*)&Qs[(c0 + 8 + j) * 72 + np2] = pkh(qa1[j], qb1[j]);
      *(uint32_t*)&Ks[(c0 + j) * 72 + np2]     = pkh(ka0[j], kb0[j]);
      *(uint32_t*)&Ks[(c0 + 8 + j) * 72 + np2] = pkh(ka1[j], kb1[j]);
    }
    __syncthreads();
#pragma unroll
    for (int kk = 0; kk < 64; kk += 32) {
      v16h a = ld_frag(&Qs[(wave * 16 + (lane & 15)) * 72 + kk + (lane >> 4) * 8]);
#pragma unroll
      for (int j = 0; j < 6; j++) {
        v16h bf = ld_frag(&Ks[(j * 16 + (lane & 15)) * 72 + kk + (lane >> 4) * 8]);
        acc[j] = wmma16(a, bf, acc[j]);
      }
    }
    __syncthreads();
  }

#pragma unroll
  for (int j = 0; j < 6; j++) {
#pragma unroll
    for (int e = 0; e < 8; e++) {
      int d = wave * 16 + (lane >> 4) * 8 + e;
      int ec = j * 16 + (lane & 15);
      S[((size_t)bh * NHD + d) * NHD + ec] = acc[j][e];
    }
  }
}

// ---------------------------------------------------------------------------
// Softmax over e with the deferred cosine norms:
//   attn[d][e] = softmax_e( S[d][e] * qinv[d] * kinv[e] * SCALE )  -> f16
// ---------------------------------------------------------------------------
__global__ void softmax_kernel(const float* __restrict__ S,
                               const float* __restrict__ qinv,
                               const float* __restrict__ kinv,
                               _Float16* __restrict__ attn) {
  int bh = blockIdx.x;
  int b = bh >> 3, h = bh & 7;
  int off = h * NHD;
  int d = threadIdx.x;  // blockDim == 96
  const float* srow = S + ((size_t)bh * NHD + d) * NHD;
  float qf = qinv[b * NC + off + d] * SCALE;
  float vals[NHD];
  float mx = -3.4e38f;
#pragma unroll 8
  for (int e = 0; e < NHD; e++) {
    float v = srow[e] * qf * kinv[b * NC + off + e];
    vals[e] = v;
    mx = fmaxf(mx, v);
  }
  float sum = 0.f;
#pragma unroll 8
  for (int e = 0; e < NHD; e++) {
    vals[e] = expf(vals[e] - mx);
    sum += vals[e];
  }
  float inv = 1.f / sum;
  _Float16* arow = attn + ((size_t)bh * NHD + d) * NHD;
#pragma unroll 8
  for (int e = 0; e < NHD; e++) arow[e] = (_Float16)(vals[e] * inv);
}

// ---------------------------------------------------------------------------
// X2[b][n][off+d] = sum_e V[b][n][768+off+e] * attn[bh][d][e]   (K = 96)
// grid (MTOK/128, H); 8 waves of 16 tokens; A frags loaded straight from
// global (row-major V); attn tile (18KB f16) brought into LDS with
// CDNA5 async global->LDS copies (ASYNCcnt), fragments read via ds_load_b128.
// ---------------------------------------------------------------------------
__global__ void __launch_bounds__(256)
av_kernel(const _Float16* __restrict__ KV, const _Float16* __restrict__ attn,
          _Float16* __restrict__ X2) {
  __shared__ _Float16 At[NHD * NHD];   // attn[d][e], 18432 bytes
  int h = blockIdx.y;
  int off = h * NHD;
  int n0 = blockIdx.x * 128;           // flat token base (whole block same batch)
  int b = n0 >> 12;
  int t = threadIdx.x, lane = t & 31, wave = t >> 5;  // 8 waves

  // ---- async copy attn[bh] (96x96 f16 = 1152 x b128) into LDS ----
  // threads 0..191 issue 6 x 16B transfers each; offset applies to both sides.
  if (t < 192) {
    const char* gsrc =
        (const char*)(attn + (size_t)(b * NH + h) * NHD * NHD) + t * 96;
    uint32_t lds = (uint32_t)(uintptr_t)(&At[0]) + t * 96;
#pragma unroll
    for (int i = 0; i < 6; i++) {
      asm volatile("global_load_async_to_lds_b128 %0, %1, off offset:%2"
                   :: "v"(lds), "v"(gsrc), "i"(i * 16)
                   : "memory");
    }
  }
  asm volatile("s_wait_asynccnt 0x0" ::: "memory");
  __syncthreads();

  int row = n0 + wave * 16 + (lane & 15);

  v8f acc[6];
#pragma unroll
  for (int j = 0; j < 6; j++) acc[j] = v8f_zero();

  const _Float16* Abase = KV + (size_t)row * (2 * NC) + NC + off;

#pragma unroll
  for (int kk = 0; kk < NHD; kk += 32) {
    v16h a = ld_frag(Abase + kk + (lane >> 4) * 8);
#pragma unroll
    for (int j = 0; j < 6; j++) {
      v16h bf = ld_frag(&At[(size_t)(j * 16 + (lane & 15)) * NHD + kk + (lane >> 4) * 8]);
      acc[j] = wmma16(a, bf, acc[j]);
    }
  }

#pragma unroll
  for (int j = 0; j < 6; j++) {
#pragma unroll
    for (int e = 0; e < 8; e++) {
      int tok = n0 + wave * 16 + (lane >> 4) * 8 + e;
      X2[(size_t)tok * NC + off + j * 16 + (lane & 15)] = (_Float16)acc[j][e];
    }
  }
}

// ---------------------------------------------------------------------------
extern "C" void kernel_launch(void* const* d_in, const int* in_sizes, int n_in,
                              void* d_out, int out_size, void* d_ws, size_t ws_size,
                              hipStream_t stream) {
  (void)in_sizes; (void)n_in; (void)out_size; (void)ws_size;
  const float* Xc    = (const float*)d_in[0];
  const float* Xd    = (const float*)d_in[1];
  const float* Wq    = (const float*)d_in[2];
  const float* Wkv   = (const float*)d_in[3];
  const float* Wproj = (const float*)d_in[4];
  const float* bproj = (const float*)d_in[5];

  char* ws = (char*)d_ws;
  const size_t OFF_Q    = 0;                                 // 16384*768*2
  const size_t OFF_KV   = OFF_Q + (size_t)MTOK * NC * 2;     // 16384*1536*2
  const size_t OFF_S    = OFF_KV + (size_t)MTOK * 2 * NC * 2;
  const size_t OFF_ATTN = OFF_S + (size_t)NB * NH * NHD * NHD * 4;
  const size_t OFF_QN   = OFF_ATTN + (size_t)NB * NH * NHD * NHD * 2;
  const size_t OFF_KN   = OFF_QN + (size_t)NB * NC * 4;

  _Float16* Q    = (_Float16*)(ws + OFF_Q);
  _Float16* KV   = (_Float16*)(ws + OFF_KV);
  float*    S    = (float*)(ws + OFF_S);
  _Float16* attn = (_Float16*)(ws + OFF_ATTN);
  float*    qn   = (float*)(ws + OFF_QN);
  float*    kn   = (float*)(ws + OFF_KN);
  _Float16* X2   = (_Float16*)(ws + OFF_Q);  // alias: Q dead after s_kernel

  // 1) Q = Xc @ Wq        [16384 x 768]
  gemm_kernel<false, false><<<dim3(NC / 128, MTOK / 128), 256, 0, stream>>>(
      Xc, Wq, Q, nullptr, MTOK, NC, NC);
  // 2) KV = Xd @ Wkv      [16384 x 1536]
  gemm_kernel<false, false><<<dim3(2 * NC / 128, MTOK / 128), 256, 0, stream>>>(
      Xd, Wkv, KV, nullptr, MTOK, 2 * NC, NC);
  // 3) token-axis L2 norms (inverse) for q and k
  hipMemsetAsync(qn, 0, 2 * (size_t)NB * NC * sizeof(float), stream);
  sumsq_kernel<<<dim3(NB, NC / 128, 16), 128, 0, stream>>>(Q, NC, qn);
  sumsq_kernel<<<dim3(NB, NC / 128, 16), 128, 0, stream>>>(KV, 2 * NC, kn);
  invnorm_kernel<<<(2 * NB * NC + 255) / 256, 256, 0, stream>>>(qn, 2 * NB * NC);
  // 4) S = Q_h^T K_h per (b,h)   [32 x 96 x 96]
  s_kernel<<<NB * NH, 192, 0, stream>>>(Q, KV, S);
  // 5) attn = softmax(S * qinv * kinv * scale)
  softmax_kernel<<<NB * NH, NHD, 0, stream>>>(S, qn, kn, attn);
  // 6) X2 = V @ attn^T per head   [16384 x 768]
  av_kernel<<<dim3(MTOK / 128, NH), 256, 0, stream>>>(KV, attn, X2);
  // 7) out = X2 @ Wproj + bproj   (f32)
  gemm_kernel<true, true><<<dim3(NC / 128, MTOK / 128), 256, 0, stream>>>(
      X2, Wproj, d_out, bproj, MTOK, NC, NC);
}
